// HyperbolicFlashAttention_43224550866961
// MI455X (gfx1250) — matrix operand order repaired and checked by reading, compiled, not run on hardware
//
#include <hip/hip_runtime.h>
#include <hip/hip_bf16.h>

typedef __attribute__((ext_vector_type(16))) __bf16 v16bf;
typedef __attribute__((ext_vector_type(8)))  __bf16 v8bf;
typedef __attribute__((ext_vector_type(8)))  float  v8f;

#define B_   2
#define N_   2048
#define D_   768
#define H_   12
#define HD_  64
#define SCALE_ 0.125f      /* HD^-0.5 */
#define EPS_ 1e-8f

// ---------------------------------------------------------------------------
// WMMA helpers (CDNA5 wave32, 16x16x32 bf16 -> f32)
// ---------------------------------------------------------------------------
__device__ __forceinline__ v8f wmma_bf16(v16bf a, v16bf b, v8f c) {
  return __builtin_amdgcn_wmma_f32_16x16x32_bf16(false, a, false, b, (short)0, c,
                                                 false, false);
}

// A-matrix 16x32 bf16 fragment: lane = khalf*16 + M,
// per-lane elements i: K = khalf*16 + i   (two 16B LDS reads)
__device__ __forceinline__ v16bf load_a_frag(const __bf16* row, int khalf) {
  v8bf lo = *(const v8bf*)(row + khalf * 16);
  v8bf hg = *(const v8bf*)(row + khalf * 16 + 8);
  v16bf r;
#pragma unroll
  for (int i = 0; i < 8; ++i) { r[i] = lo[i]; r[8 + i] = hg[i]; }
  return r;
}

// B-matrix 32x16 bf16 fragment: lane = khalf*16 + N, elements K = khalf*16 + i
__device__ __forceinline__ v16bf load_b_frag(const __bf16* col, int khalf) {
  v8bf lo = *(const v8bf*)(col + khalf * 16);
  v8bf hg = *(const v8bf*)(col + khalf * 16 + 8);
  v16bf r;
#pragma unroll
  for (int i = 0; i < 8; ++i) { r[i] = lo[i]; r[8 + i] = hg[i]; }
  return r;
}

// CDNA5 async global->LDS copy (16B per lane), tracked by ASYNCcnt.
__device__ __forceinline__ void async_load_b128(unsigned lds_off, const void* gaddr) {
  asm volatile("global_load_async_to_lds_b128 %0, %1, off"
               :: "v"(lds_off), "v"(gaddr) : "memory");
}
__device__ __forceinline__ void wait_async_le(int) {}  // doc helper only
#define WAIT_ASYNC_4() asm volatile("s_wait_asynccnt 0x4" ::: "memory")
#define WAIT_ASYNC_0() asm volatile("s_wait_asynccnt 0x0" ::: "memory")
#define WAIT_DS_0()    asm volatile("s_wait_dscnt 0x0"    ::: "memory")

__device__ __forceinline__ unsigned lds_addr32(const void* p) {
  return (unsigned)(unsigned long long)p;   // generic LDS ptr low 32b = LDS offset
}

// ---------------------------------------------------------------------------
// log0: x -> tangent vector, converted to bf16 for WMMA GEMM consumption
// ---------------------------------------------------------------------------
__global__ __launch_bounds__(256) void log0_to_bf16_kernel(
    const float* __restrict__ x, const float* __restrict__ curvature,
    __bf16* __restrict__ out) {
  __shared__ float red[256];
  const int row = blockIdx.x;
  const int tid = threadIdx.x;
  const float* xr = x + (size_t)row * D_;
  float ss = 0.f;
  for (int i = tid; i < D_; i += 256) { float v = xr[i]; ss += v * v; }
  red[tid] = ss;
  __syncthreads();
  for (int s = 128; s > 0; s >>= 1) {
    if (tid < s) red[tid] += red[tid + s];
    __syncthreads();
  }
  const float scn = sqrtf(curvature[0]);
  float n = fmaxf(sqrtf(red[0]), EPS_);
  float t = fminf(scn * n, 1.0f - 1e-7f);
  float art = 0.5f * __logf((1.0f + t) / (1.0f - t));   // artanh
  float scale = art / (scn * n);
  for (int i = tid; i < D_; i += 256)
    out[(size_t)row * D_ + i] = (__bf16)(xr[i] * scale);
}

// ---------------------------------------------------------------------------
// exp0: tangent -> manifold, fp32 output
// ---------------------------------------------------------------------------
__global__ __launch_bounds__(256) void exp0_kernel(
    const float* __restrict__ v, const float* __restrict__ curvature,
    float* __restrict__ out) {
  __shared__ float red[256];
  const int row = blockIdx.x;
  const int tid = threadIdx.x;
  const float* vr = v + (size_t)row * D_;
  float ss = 0.f;
  for (int i = tid; i < D_; i += 256) { float w = vr[i]; ss += w * w; }
  red[tid] = ss;
  __syncthreads();
  for (int s = 128; s > 0; s >>= 1) {
    if (tid < s) red[tid] += red[tid + s];
    __syncthreads();
  }
  const float scn = sqrtf(curvature[0]);
  float n = fmaxf(sqrtf(red[0]), EPS_);
  float scale = tanhf(scn * n) / (scn * n);
  for (int i = tid; i < D_; i += 256)
    out[(size_t)row * D_ + i] = vr[i] * scale;
}

// ---------------------------------------------------------------------------
__global__ void f32_to_bf16_kernel(const float* __restrict__ in,
                                   __bf16* __restrict__ out, int n) {
  int i = blockIdx.x * blockDim.x + threadIdx.x;
  int stride = gridDim.x * blockDim.x;
  for (; i < n; i += stride) out[i] = (__bf16)in[i];
}

// ---------------------------------------------------------------------------
// C[M,E] = A[M,K] * B[E,K]^T (+bias), bf16 inputs, fp32 out.
// 128x128 block tile, BK=32, 8 waves each computing 32x64 (8 WMMA tiles).
// Double-buffered tile staging via global_load_async_to_lds_b128 (ASYNCcnt).
// ---------------------------------------------------------------------------
__global__ __launch_bounds__(256) void gemm_nt_bf16_kernel(
    const __bf16* __restrict__ A, const __bf16* __restrict__ Bm,
    const float* __restrict__ bias, float* __restrict__ C,
    int M, int E, int K) {
  __shared__ __align__(16) __bf16 As[2][128][32];
  __shared__ __align__(16) __bf16 Bs[2][128][32];
  const int tid  = threadIdx.x;
  const int wave = tid >> 5;
  const int lane = tid & 31;
  const int hi   = lane >> 4;
  const int ln   = lane & 15;
  const int m0 = blockIdx.x * 128;
  const int e0 = blockIdx.y * 128;
  const int wm = (wave & 3) * 32;
  const int we = (wave >> 2) * 64;

  v8f acc[2][4] = {};
  const int lr = tid >> 1;             // 0..127: tile row loaded by this lane
  const int ls = (tid & 1) * 16;       // 16-element segment within BK=32

  const __bf16* gA = &A[(size_t)(m0 + lr) * K + ls];
  const __bf16* gB = &Bm[(size_t)(e0 + lr) * K + ls];
  const unsigned ldsA0 = lds_addr32(&As[0][lr][ls]);
  const unsigned ldsA1 = lds_addr32(&As[1][lr][ls]);
  const unsigned ldsB0 = lds_addr32(&Bs[0][lr][ls]);
  const unsigned ldsB1 = lds_addr32(&Bs[1][lr][ls]);

  auto issue = [&](int buf, int k0) {
    unsigned la = buf ? ldsA1 : ldsA0;
    unsigned lb = buf ? ldsB1 : ldsB0;
    async_load_b128(la,      gA + k0);
    async_load_b128(la + 16, gA + k0 + 8);
    async_load_b128(lb,      gB + k0);
    async_load_b128(lb + 16, gB + k0 + 8);
  };

  const int nk = K / 32;
  issue(0, 0);
  for (int it = 0; it < nk; ++it) {
    if (it + 1 < nk) {                  // prefetch next tile into other buffer
      issue((it + 1) & 1, (it + 1) * 32);
      WAIT_ASYNC_4();                   // in-order: current group complete
    } else {
      WAIT_ASYNC_0();
    }
    __syncthreads();                    // all waves' tiles landed

    const int buf = it & 1;
    v16bf af[2], bfr[4];
#pragma unroll
    for (int mi = 0; mi < 2; ++mi)
      af[mi] = load_a_frag(&As[buf][wm + mi * 16 + ln][0], hi);
#pragma unroll
    for (int ei = 0; ei < 4; ++ei)
      bfr[ei] = load_b_frag(&Bs[buf][we + ei * 16 + ln][0], hi);
#pragma unroll
    for (int mi = 0; mi < 2; ++mi)
#pragma unroll
      for (int ei = 0; ei < 4; ++ei)
        acc[mi][ei] = wmma_bf16(af[mi], bfr[ei], acc[mi][ei]);

    __syncthreads();                    // reads done before buffer reuse
  }

#pragma unroll
  for (int mi = 0; mi < 2; ++mi)
#pragma unroll
    for (int ei = 0; ei < 4; ++ei) {
      int col = e0 + we + ei * 16 + ln;
      float bv = bias ? bias[col] : 0.0f;
#pragma unroll
      for (int j = 0; j < 8; ++j) {
        int row = m0 + wm + mi * 16 + j + 8 * hi;  // C layout: M = j + 8*(lane>=16)
        C[(size_t)row * E + col] = acc[mi][ei][j] + bv;
      }
    }
}

// ---------------------------------------------------------------------------
// Hyperbolic flash attention, one (b,h,64-query-block) per workgroup, 4 waves.
// exp0 fused into Q/K tile loads; ||exp0(v)||^2 = (tanh(sc*n)/sc)^2 closed form.
// ---------------------------------------------------------------------------
__global__ __launch_bounds__(128) void hyp_attn_kernel(
    const float* __restrict__ qkv, const float* __restrict__ curvature,
    __bf16* __restrict__ attn_out) {
  __shared__ __align__(16) __bf16 Qs[64][64];       // q on manifold, bf16
  __shared__ __align__(16) __bf16 Ks[64][64];       // k on manifold, bf16
  __shared__ __align__(16) __bf16 Vt[64][64];       // V transposed [hd][key]
  __shared__ __align__(16) __bf16 Ps[4][16][64];    // probs, per-wave region
  __shared__ float qns[64], qaf[64], kns[64], kaf[64];

  const float sc = sqrtf(curvature[0]);
  const int bh = blockIdx.y;
  const int b  = bh / H_;
  const int h  = bh % H_;
  const int q0 = blockIdx.x * 64;
  const int tid  = threadIdx.x;
  const int wave = tid >> 5;
  const int lane = tid & 31;
  const int hi   = lane >> 4;
  const int ln   = lane & 15;

  { // ---- Q block: exp0 -> manifold, bf16 to LDS (2 threads per row)
    const int r = tid >> 1;
    const int s = (tid & 1) * 32;
    const float* src = qkv + (size_t)(b * N_ + q0 + r) * (3 * D_) + h * HD_ + s;
    float xb[32]; float ss = 0.f;
#pragma unroll
    for (int i = 0; i < 32; ++i) { xb[i] = src[i]; ss += xb[i] * xb[i]; }
    ss += __shfl_xor(ss, 1, 32);
    float n  = fmaxf(sqrtf(ss), EPS_);
    float th = tanhf(sc * n);
    float t  = th / (sc * n);
#pragma unroll
    for (int i = 0; i < 32; ++i) Qs[r][s + i] = (__bf16)(xb[i] * t);
    if (s == 0) {
      float q2 = (th / sc) * (th / sc);
      qns[r] = q2;
      qaf[r] = 1.0f - fminf(q2, 0.99f);
    }
  }
  __syncthreads();

  float q_n[8], q_a[8];
#pragma unroll
  for (int j = 0; j < 8; ++j) {
    int r = wave * 16 + j + 8 * hi;
    q_n[j] = qns[r]; q_a[j] = qaf[r];
  }

  v8f o[4] = {};
  float mrow[8], lrow[8];
#pragma unroll
  for (int j = 0; j < 8; ++j) { mrow[j] = -__builtin_inff(); lrow[j] = 0.f; }

  for (int kb = 0; kb < N_; kb += 64) {
    __syncthreads();
    { // ---- K block exp0 -> Ks
      const int r = tid >> 1;
      const int s = (tid & 1) * 32;
      const float* src = qkv + (size_t)(b * N_ + kb + r) * (3 * D_) + D_ + h * HD_ + s;
      float xb[32]; float ss = 0.f;
#pragma unroll
      for (int i = 0; i < 32; ++i) { xb[i] = src[i]; ss += xb[i] * xb[i]; }
      ss += __shfl_xor(ss, 1, 32);
      float n  = fmaxf(sqrtf(ss), EPS_);
      float th = tanhf(sc * n);
      float t  = th / (sc * n);
#pragma unroll
      for (int i = 0; i < 32; ++i) Ks[r][s + i] = (__bf16)(xb[i] * t);
      if (s == 0) {
        float k2 = (th / sc) * (th / sc);
        kns[r] = k2;
        kaf[r] = 1.0f - fminf(k2, 0.99f);
      }
    }
    { // ---- V block -> Vt (transposed so B-fragments are key-contiguous)
      const int r = tid >> 1;
      const int s = (tid & 1) * 32;
      const float* src = qkv + (size_t)(b * N_ + kb + r) * (3 * D_) + 2 * D_ + h * HD_ + s;
#pragma unroll
      for (int i = 0; i < 32; ++i) Vt[s + i][r] = (__bf16)src[i];
    }
    __syncthreads();

    // ---- scores S = Qm * Km^T, two K=32 WMMA steps over HD=64
    v16bf aq0 = load_a_frag(&Qs[wave * 16 + ln][0], hi);
    v16bf aq1 = load_a_frag(&Qs[wave * 16 + ln][32], hi);

    float sv[4][8];
    float rmax[8];
#pragma unroll
    for (int j = 0; j < 8; ++j) rmax[j] = -__builtin_inff();

#pragma unroll
    for (int t = 0; t < 4; ++t) {
      const int key = t * 16 + ln;
      const float k_n = kns[key];
      const float k_a = kaf[key];
      v16bf bk0 = load_b_frag(&Ks[key][0], hi);
      v16bf bk1 = load_b_frag(&Ks[key][32], hi);
      v8f acc = {};
      acc = wmma_bf16(aq0, bk0, acc);
      acc = wmma_bf16(aq1, bk1, acc);
#pragma unroll
      for (int j = 0; j < 8; ++j) {
        float dot   = acc[j];
        float diff  = fmaxf(q_n[j] + k_n - 2.0f * dot, 0.0f);
        float den   = q_a[j] * k_a + 1e-8f;
        float inner = fmaxf(1.0f + 2.0f * diff / den, 1.0f + 1e-7f);
        float dist  = __logf(inner + sqrtf(inner * inner - 1.0f)); // arccosh
        float s     = -dist * SCALE_;
        sv[t][j] = s;
        rmax[j]  = fmaxf(rmax[j], s);
      }
    }
    // row-max across the 16 lanes sharing a row group (xor masks stay in-half)
#pragma unroll
    for (int m = 1; m < 16; m <<= 1)
#pragma unroll
      for (int j = 0; j < 8; ++j)
        rmax[j] = fmaxf(rmax[j], __shfl_xor(rmax[j], m, 32));

    float alpha[8], rsum[8];
#pragma unroll
    for (int j = 0; j < 8; ++j) {
      float mn = fmaxf(mrow[j], rmax[j]);
      alpha[j] = __expf(mrow[j] - mn);
      mrow[j]  = mn;
      rsum[j]  = 0.f;
    }
#pragma unroll
    for (int t = 0; t < 4; ++t)
#pragma unroll
      for (int j = 0; j < 8; ++j) {
        float p = __expf(sv[t][j] - mrow[j]);
        rsum[j] += p;
        Ps[wave][j + 8 * hi][t * 16 + ln] = (__bf16)p;
      }
#pragma unroll
    for (int m = 1; m < 16; m <<= 1)
#pragma unroll
      for (int j = 0; j < 8; ++j)
        rsum[j] += __shfl_xor(rsum[j], m, 32);
#pragma unroll
    for (int j = 0; j < 8; ++j) lrow[j] = lrow[j] * alpha[j] + rsum[j];
#pragma unroll
    for (int t2 = 0; t2 < 4; ++t2)
#pragma unroll
      for (int j = 0; j < 8; ++j) o[t2][j] *= alpha[j];

    // Ps is wave-private: wave-local LDS drain instead of workgroup barrier
    WAIT_DS_0();

    // ---- O += P * V, K = 64 keys in two WMMA steps
    v16bf ap0 = load_a_frag(&Ps[wave][ln][0], hi);
    v16bf ap1 = load_a_frag(&Ps[wave][ln][32], hi);
#pragma unroll
    for (int t2 = 0; t2 < 4; ++t2) {
      v16bf bv0 = load_b_frag(&Vt[t2 * 16 + ln][0], hi);
      v16bf bv1 = load_b_frag(&Vt[t2 * 16 + ln][32], hi);
      o[t2] = wmma_bf16(ap0, bv0, o[t2]);
      o[t2] = wmma_bf16(ap1, bv1, o[t2]);
    }
  }

  // ---- epilogue: normalize, store bf16 into [B*N, D] layout (h*HD + hd)
#pragma unroll
  for (int t2 = 0; t2 < 4; ++t2)
#pragma unroll
    for (int j = 0; j < 8; ++j) {
      int row = q0 + wave * 16 + j + 8 * hi;
      int col = h * HD_ + t2 * 16 + ln;
      attn_out[(size_t)(b * N_ + row) * D_ + col] = (__bf16)(o[t2][j] / lrow[j]);
    }
}

// ---------------------------------------------------------------------------
extern "C" void kernel_launch(void* const* d_in, const int* in_sizes, int n_in,
                              void* d_out, int out_size, void* d_ws, size_t ws_size,
                              hipStream_t stream) {
  (void)in_sizes; (void)n_in; (void)out_size; (void)ws_size;
  const float* x      = (const float*)d_in[0];
  const float* w_qkv  = (const float*)d_in[1];
  const float* w_proj = (const float*)d_in[2];
  const float* b_proj = (const float*)d_in[3];
  const float* curv   = (const float*)d_in[4];

  char* ws = (char*)d_ws;
  size_t off = 0;
  auto take = [&](size_t bytes) {
    char* p = ws + off;
    off = (off + bytes + 255) & ~(size_t)255;
    return p;
  };
  __bf16* vbf      = (__bf16*)take((size_t)B_ * N_ * D_ * 2);       // log0(x) bf16
  __bf16* wqkv_bf  = (__bf16*)take((size_t)3 * D_ * D_ * 2);
  __bf16* wproj_bf = (__bf16*)take((size_t)D_ * D_ * 2);
  float*  qkv      = (float*)take((size_t)B_ * N_ * 3 * D_ * 4);    // fp32 qkv
  __bf16* attn_bf  = (__bf16*)take((size_t)B_ * N_ * D_ * 2);       // attn out bf16
  float*  vout     = (float*)take((size_t)B_ * N_ * D_ * 4);        // pre-exp0 fp32

  log0_to_bf16_kernel<<<B_ * N_, 256, 0, stream>>>(x, curv, vbf);
  f32_to_bf16_kernel<<<512, 256, 0, stream>>>(w_qkv, wqkv_bf, 3 * D_ * D_);
  f32_to_bf16_kernel<<<512, 256, 0, stream>>>(w_proj, wproj_bf, D_ * D_);

  // qkv = v @ w_qkv^T : M=4096, E=2304, K=768
  gemm_nt_bf16_kernel<<<dim3((B_ * N_) / 128, (3 * D_) / 128), 256, 0, stream>>>(
      vbf, wqkv_bf, nullptr, qkv, B_ * N_, 3 * D_, D_);

  // flash attention: grid (query blocks, B*H)
  hyp_attn_kernel<<<dim3(N_ / 64, B_ * H_), 128, 0, stream>>>(qkv, curv, attn_bf);

  // vout = attn @ w_proj^T + b_proj : M=4096, E=768, K=768
  gemm_nt_bf16_kernel<<<dim3((B_ * N_) / 128, D_ / 128), 256, 0, stream>>>(
      attn_bf, wproj_bf, b_proj, vout, B_ * N_, D_, D_);

  exp0_kernel<<<B_ * N_, 256, 0, stream>>>(vout, curv, (float*)d_out);
}
